// HexGoNet_22136261443950
// MI455X (gfx1250) — compile-verified, matrix operand order checked by reading, not run on hardware
//
#include <hip/hip_runtime.h>
#include <hip/hip_bf16.h>
#include <math.h>

// ---------------------------------------------------------------------------
// Types for CDNA5 WMMA (gfx1250, wave32)
// ---------------------------------------------------------------------------
typedef __attribute__((ext_vector_type(16))) __bf16 v16bf;
typedef __attribute__((ext_vector_type(2)))  __bf16 bf16x2;
typedef __attribute__((ext_vector_type(8)))  float  v8f;

#define NN   10000
#define EE   60000
#define ETOT 70000   // EE + NN self loops
#define HDIM 256
#define HEADS 4
#define CDIM 64
#define NBLK 8
#define BN_EPS 1e-5f

// activation helpers --------------------------------------------------------
__device__ __forceinline__ float mishf(float x) {
    float sp = log1pf(__expf(x));          // softplus
    return x * tanhf(sp);
}
__device__ __forceinline__ unsigned enc_f(float x) {
    unsigned u = __float_as_uint(x);
    return (u & 0x80000000u) ? ~u : (u | 0x80000000u);   // order-preserving
}
__device__ __forceinline__ float dec_f(unsigned u) {
    return (u & 0x80000000u) ? __uint_as_float(u & 0x7fffffffu)
                             : __uint_as_float(~u);
}
#define ENC_NEG_INF 0x007fffffu   // enc_f(-inf)

// ---------------------------------------------------------------------------
// Weight convert+transpose: W fp32 [K x N] -> Wt bf16 [N x K]
// ---------------------------------------------------------------------------
__global__ void k_wt(const float* __restrict__ W, __bf16* __restrict__ Wt,
                     int K, int Nout) {
    int idx = blockIdx.x * blockDim.x + threadIdx.x;
    if (idx >= K * Nout) return;
    int n = idx / K, k = idx % K;
    Wt[idx] = (__bf16)W[(size_t)k * Nout + n];
}

// ---------------------------------------------------------------------------
// WMMA bf16 GEMM, K fixed = 256, all-bf16 operands.
// Y[M x Nout] = Abf[M x 256] @ Wt^T   (Wt bf16 [Nout x 256]) (+ bias)
// grid: (ceil(M/64), Nout/(16*NT)), block 128 (4 waves).
// Wave w handles rows [blk.x*64 + w*16, +16), cols [blk.y*16*NT, +16*NT)
// with NT accumulators -> A fragment reused NT times per K-slab.
// All NT B fragments are loaded before the first consuming WMMA so the
// in-order LOADcnt lets WMMAs overlap the remaining loads in flight.
// Fragment layouts per CDNA5 ISA 7.12.2:
//   A (16x32 bf16): lane l row = l&15; element e -> K = e%8 + (e/8)*16 + (l>>4)*8
//   B (32x16 bf16): lane l col = l&15; same K mapping
//   C/D (16x16 f32): vgpr r, lane l -> M = r + (l>>4)*8, N = l&15
// ---------------------------------------------------------------------------
template <int NT>
__global__ void k_gemm_bf16(const __bf16* __restrict__ Abf,
                            const __bf16* __restrict__ Wt,
                            const float* __restrict__ bias,
                            float* __restrict__ Y,
                            int M, int Nout) {
    const int lane  = threadIdx.x & 31;
    const int wave  = threadIdx.x >> 5;
    const int row16 = blockIdx.x * 64 + wave * 16;
    const int col0  = blockIdx.y * (16 * NT);

    int r = row16 + (lane & 15);
    const int arow = (r < M) ? r : (M - 1);    // clamp loads; stores are guarded
    const int krl  = (lane >> 4) * 8;
    const __bf16* abase = Abf + (size_t)arow * 256 + krl;
    const __bf16* bbase = Wt + (size_t)(col0 + (lane & 15)) * 256 + krl;

    v8f acc[NT];
#pragma unroll
    for (int t = 0; t < NT; ++t) acc[t] = {};

#pragma unroll
    for (int k0 = 0; k0 < 256; k0 += 32) {
        const __bf16* ap = abase + k0;
        if (k0 + 32 < 256) __builtin_prefetch(ap + 32, 0, 0);
        v16bf a;
#pragma unroll
        for (int p = 0; p < 8; ++p) {
            const int ko = ((p & 3) << 1) + ((p >> 2) << 4); // 0,2,4,6,16,18,20,22
            bf16x2 w2 = *(const bf16x2*)(ap + ko);
            a[2 * p] = w2.x; a[2 * p + 1] = w2.y;
        }
        // stage ALL B fragments first (batched independent loads) ...
        v16bf bfrag[NT];
#pragma unroll
        for (int t = 0; t < NT; ++t) {
            const __bf16* bp = bbase + (size_t)t * 16 * 256 + k0;
#pragma unroll
            for (int p = 0; p < 8; ++p) {
                const int ko = ((p & 3) << 1) + ((p >> 2) << 4);
                bf16x2 w2 = *(const bf16x2*)(bp + ko);
                bfrag[t][2 * p] = w2.x; bfrag[t][2 * p + 1] = w2.y;
            }
        }
        // ... then issue the NT WMMAs (descending loadcnt waits)
#pragma unroll
        for (int t = 0; t < NT; ++t) {
            acc[t] = __builtin_amdgcn_wmma_f32_16x16x32_bf16(
                false, a, false, bfrag[t], (short)0, acc[t], false, false);
        }
    }

#pragma unroll
    for (int t = 0; t < NT; ++t) {
        const int ncol = col0 + t * 16 + (lane & 15);
        const float bj = bias ? bias[ncol] : 0.f;
#pragma unroll
        for (int rr = 0; rr < 8; ++rr) {
            int mrow = row16 + rr + ((lane >> 4) * 8);
            if (mrow < M) Y[(size_t)mrow * Nout + ncol] = acc[t][rr] + bj;
        }
    }
}

// ---------------------------------------------------------------------------
// Glue kernels (fp32)
// ---------------------------------------------------------------------------
__global__ void k_fillf(float* p, float v, long n) {
    long i = (long)blockIdx.x * blockDim.x + threadIdx.x;
    if (i < n) p[i] = v;
}
__global__ void k_fillu(unsigned* p, unsigned v, long n) {
    long i = (long)blockIdx.x * blockDim.x + threadIdx.x;
    if (i < n) p[i] = v;
}

// input transform: h = x @ W_in  (bias cancels in following BN)
__global__ void k_in(const float* __restrict__ x, const float* __restrict__ W,
                     float* __restrict__ h) {
    int i = blockIdx.x * blockDim.x + threadIdx.x;
    if (i >= NN * HDIM) return;
    int node = i >> 8, c = i & 255;
    float s = 0.f;
#pragma unroll
    for (int k = 0; k < 8; ++k) s += x[node * 8 + k] * W[k * HDIM + c];
    h[i] = s;
}

// u_emb = mish(u @ W_u + b_u), 64 wide
__global__ void k_uemb(const float* __restrict__ u, const float* __restrict__ Wu,
                       const float* __restrict__ bu, float* __restrict__ out) {
    int j = threadIdx.x;
    float s = bu[j] + u[0] * Wu[j] + u[1] * Wu[64 + j];
    out[j] = mishf(s);
}

// degree over dst (incl. self loops)
__global__ void k_deg(const int* __restrict__ ei, float* __restrict__ deg) {
    int e = blockIdx.x * blockDim.x + threadIdx.x;
    if (e >= ETOT) return;
    int dst = (e < EE) ? ei[EE + e] : (e - EE);
    atomicAdd(&deg[dst], 1.f);
}
__global__ void k_norm(const int* __restrict__ ei, const float* __restrict__ deg,
                       float* __restrict__ norm) {
    int e = blockIdx.x * blockDim.x + threadIdx.x;
    if (e >= ETOT) return;
    int s = (e < EE) ? ei[e] : (e - EE);
    int d = (e < EE) ? ei[EE + e] : (e - EE);
    norm[e] = rsqrtf(deg[s]) * rsqrtf(deg[d]);
}

// GATv2 edge logits + segment-max (encoded) -------------------------------
__global__ void k_edge_logit(const float* __restrict__ xl, const float* __restrict__ xr,
                             const float* __restrict__ att, const int* __restrict__ ei,
                             float* __restrict__ logit, unsigned* __restrict__ menc) {
    int idx = blockIdx.x * blockDim.x + threadIdx.x;
    if (idx >= ETOT * HEADS) return;
    int e = idx >> 2, h = idx & 3;
    int src = (e < EE) ? ei[e] : (e - EE);
    int dst = (e < EE) ? ei[EE + e] : (e - EE);
    const float* pl = xl + (size_t)src * HDIM + h * CDIM;
    const float* pr = xr + (size_t)dst * HDIM + h * CDIM;
    const float* pa = att + h * CDIM;
    float s = 0.f;
#pragma unroll 8
    for (int c = 0; c < CDIM; ++c) {
        float v = pl[c] + pr[c];
        v = (v > 0.f) ? v : 0.2f * v;          // leaky_relu(0.2)
        s += pa[c] * v;
    }
    logit[idx] = s;
    atomicMax(menc + dst * HEADS + h, enc_f(s));
}

__global__ void k_edge_w(const float* __restrict__ logit, const int* __restrict__ ei,
                         const unsigned* __restrict__ menc,
                         float* __restrict__ wbuf, float* __restrict__ denom) {
    int idx = blockIdx.x * blockDim.x + threadIdx.x;
    if (idx >= ETOT * HEADS) return;
    int e = idx >> 2, h = idx & 3;
    int dst = (e < EE) ? ei[EE + e] : (e - EE);
    float w = __expf(logit[idx] - dec_f(menc[dst * HEADS + h]));
    wbuf[idx] = w;
    atomicAdd(&denom[dst * HEADS + h], w);
}

// out[dst] += xl[src] * alpha   (block per edge, thread per channel)
__global__ void k_gat_scatter(const float* __restrict__ xl, const int* __restrict__ ei,
                              const float* __restrict__ wbuf, const float* __restrict__ denom,
                              float* __restrict__ out) {
    int e = blockIdx.x, ch = threadIdx.x, h = ch >> 6;
    int src = (e < EE) ? ei[e] : (e - EE);
    int dst = (e < EE) ? ei[EE + e] : (e - EE);
    float alpha = wbuf[e * HEADS + h] / (denom[dst * HEADS + h] + 1e-16f);
    atomicAdd(&out[(size_t)dst * HDIM + ch], xl[(size_t)src * HDIM + ch] * alpha);
}

// GCN scatter: out[dst][f] += tmp[src][f] * norm[e]
__global__ void k_gcn_scatter(const float* __restrict__ tmp, const int* __restrict__ ei,
                              const float* __restrict__ norm, float* __restrict__ out, int F) {
    int e = blockIdx.x, f = threadIdx.x;
    int src = (e < EE) ? ei[e] : (e - EE);
    int dst = (e < EE) ? ei[EE + e] : (e - EE);
    atomicAdd(&out[(size_t)dst * F + f], tmp[(size_t)src * F + f] * norm[e]);
}

// per-channel mean/var (biased) over M rows; one block per channel
__global__ void k_colstats(const float* __restrict__ X, int M, int C,
                           float* __restrict__ mean, float* __restrict__ var) {
    int c = blockIdx.x, t = threadIdx.x;
    __shared__ float sh[256], sh2[256];
    float s = 0.f, ss = 0.f;
    for (int i = t; i < M; i += 256) {
        float v = X[(size_t)i * C + c];
        s += v; ss += v * v;
    }
    sh[t] = s; sh2[t] = ss;
    __syncthreads();
    for (int o = 128; o > 0; o >>= 1) {
        if (t < o) { sh[t] += sh[t + o]; sh2[t] += sh2[t + o]; }
        __syncthreads();
    }
    if (t == 0) {
        float m = sh[0] / (float)M;
        mean[c] = m;
        var[c]  = sh2[0] / (float)M - m * m;
    }
}

// BN apply + activation. mode: 0 mish, 1 relu, 2 none, 3 (+res then mish)
// Optionally dual-stores the result as bf16 (feeds the next WMMA GEMM).
__global__ void k_bn_act(const float* __restrict__ X, const float* __restrict__ mean,
                         const float* __restrict__ var, const float* __restrict__ g,
                         const float* __restrict__ b, const float* __restrict__ res,
                         float* __restrict__ out, __bf16* __restrict__ obf,
                         long total, int C, int mode) {
    long i = (long)blockIdx.x * blockDim.x + threadIdx.x;
    if (i >= total) return;
    int c = (int)(i % C);
    float v = (X[i] - mean[c]) * rsqrtf(var[c] + BN_EPS) * g[c] + b[c];
    if (mode == 3) v = mishf(v + res[i]);
    else if (mode == 0) v = mishf(v);
    else if (mode == 1) v = fmaxf(v, 0.f);
    out[i] = v;
    if (obf) obf[i] = (__bf16)v;
}

// column sums * scale (pooling); one block per channel
__global__ void k_colsum(const float* __restrict__ X, int M, int C,
                         float* __restrict__ out, float scale) {
    int c = blockIdx.x, t = threadIdx.x;
    __shared__ float sh[256];
    float s = 0.f;
    for (int i = t; i < M; i += 256) s += X[(size_t)i * C + c];
    sh[t] = s;
    __syncthreads();
    for (int o = 128; o > 0; o >>= 1) {
        if (t < o) sh[t] += sh[t + o];
        __syncthreads();
    }
    if (t == 0) out[c] = sh[0] * scale;
}

// relu( [inA(Ka) ++ inB(Kb)] @ W + b ), out Fo wide (single block)
__global__ void k_dense2_relu(const float* __restrict__ inA, int Ka,
                              const float* __restrict__ inB, int Kb,
                              const float* __restrict__ W, const float* __restrict__ b,
                              float* __restrict__ out, int Fo) {
    int j = threadIdx.x;
    if (j >= Fo) return;
    float s = b[j];
    for (int k = 0; k < Ka; ++k) s += inA[k] * W[k * Fo + j];
    for (int k = 0; k < Kb; ++k) s += inB[k] * W[(Ka + k) * Fo + j];
    out[j] = fmaxf(s, 0.f);
}

// per-node projection to scalar: out[i] = act(dot(X[i,:C], w) + b)
__global__ void k_node_dot(const float* __restrict__ X, int C, const float* __restrict__ w,
                           const float* __restrict__ b, float* __restrict__ out,
                           int do_tanh) {
    int i = blockIdx.x * blockDim.x + threadIdx.x;
    if (i >= NN) return;
    float s = b[0];
    for (int k = 0; k < C; ++k) s += X[(size_t)i * C + k] * w[k];
    out[i] = do_tanh ? tanhf(s) : s;
}

__global__ void k_value_final(const float* __restrict__ vh, const float* __restrict__ Wv2,
                              const float* __restrict__ bv2, float* __restrict__ out) {
    float s = bv2[0];
    for (int k = 0; k < 256; ++k) s += vh[k] * Wv2[k];
    out[0] = tanhf(s);
}
__global__ void k_score_final(const float* __restrict__ sh, const float* __restrict__ Wsm,
                              const float* __restrict__ bsm, const float* __restrict__ Wss,
                              const float* __restrict__ bss, float* __restrict__ outm,
                              float* __restrict__ outs) {
    float a = bsm[0], b = bss[0];
    for (int k = 0; k < 128; ++k) { a += sh[k] * Wsm[k]; b += sh[k] * Wss[k]; }
    outm[0] = a;
    outs[0] = log1pf(__expf(b)) + 0.001f;      // softplus + 0.001
}

// ---------------------------------------------------------------------------
// Host orchestration
// ---------------------------------------------------------------------------
extern "C" void kernel_launch(void* const* d_in, const int* in_sizes, int n_in,
                              void* d_out, int out_size, void* d_ws, size_t ws_size,
                              hipStream_t stream) {
    (void)in_sizes; (void)n_in; (void)out_size; (void)ws_size;

    // ---- inputs (setup_inputs dict order, params flattened in insertion order)
    const float* x      = (const float*)d_in[0];
    const int*   ei     = (const int*  )d_in[1];
    const float* u      = (const float*)d_in[2];
    /* d_in[3] batch: single graph, unused */
    const float* W_in   = (const float*)d_in[4];
    /* b_in (5) cancels in BN */
    const float* g_in   = (const float*)d_in[6];
    const float* be_in  = (const float*)d_in[7];
    const float* W_u    = (const float*)d_in[8];
    const float* b_u    = (const float*)d_in[9];
    const float* Wl_all = (const float*)d_in[10];
    const float* bl_all = (const float*)d_in[11];
    const float* Wr_all = (const float*)d_in[12];
    const float* br_all = (const float*)d_in[13];
    const float* att_all= (const float*)d_in[14];
    /* cb (15) cancels in BN */
    const float* g_all  = (const float*)d_in[16];
    const float* be_all = (const float*)d_in[17];
    const float* Wp  = (const float*)d_in[18];
    /* bp (19) cancels */
    const float* gp  = (const float*)d_in[20];
    const float* bep = (const float*)d_in[21];
    const float* Wpf = (const float*)d_in[22];
    const float* bpf = (const float*)d_in[23];
    const float* Wv  = (const float*)d_in[24];
    const float* gv  = (const float*)d_in[26];
    const float* bev = (const float*)d_in[27];
    const float* Wv1 = (const float*)d_in[28];
    const float* bv1 = (const float*)d_in[29];
    const float* Wv2 = (const float*)d_in[30];
    const float* bv2 = (const float*)d_in[31];
    const float* Wo  = (const float*)d_in[32];
    const float* go  = (const float*)d_in[34];
    const float* beo = (const float*)d_in[35];
    const float* Wof = (const float*)d_in[36];
    const float* bof = (const float*)d_in[37];
    const float* Ws  = (const float*)d_in[38];
    const float* gs  = (const float*)d_in[40];
    const float* bes = (const float*)d_in[41];
    const float* Ws1 = (const float*)d_in[42];
    const float* bs1 = (const float*)d_in[43];
    const float* Wsm = (const float*)d_in[44];
    const float* bsm = (const float*)d_in[45];
    const float* Wss = (const float*)d_in[46];
    const float* bss = (const float*)d_in[47];

    float* out = (float*)d_out;
    float* out_policy = out;            // [NN]
    float* out_value  = out + NN;       // [1]
    float* out_own    = out + NN + 1;   // [NN]
    float* out_smean  = out + 2 * NN + 1;
    float* out_sstd   = out + 2 * NN + 2;

    // ---- workspace bump allocator
    char*  ws  = (char*)d_ws;
    size_t off = 0;
    auto alloc = [&](size_t bytes) -> void* {
        void* p = ws + off;
        off = (off + bytes + 255) & ~(size_t)255;
        return p;
    };
    float*    hbuf  = (float*)alloc((size_t)NN * HDIM * 4);   // residual stream (fp32)
    float*    t0    = (float*)alloc((size_t)NN * HDIM * 4);   // xl / gemm tmp
    float*    t1    = (float*)alloc((size_t)NN * HDIM * 4);   // xr / gcn out
    float*    t2    = (float*)alloc((size_t)NN * HDIM * 4);   // gat out / head act
    __bf16*   hbf   = (__bf16*)alloc((size_t)NN * HDIM * 2);  // bf16 of residual stream
    __bf16*   tbf   = (__bf16*)alloc((size_t)NN * HDIM * 2);  // bf16 of intra-block act
    float*    meanb = (float*)alloc(HDIM * 4);
    float*    varb  = (float*)alloc(HDIM * 4);
    float*    logit = (float*)alloc((size_t)ETOT * HEADS * 4);
    float*    wbuf  = (float*)alloc((size_t)ETOT * HEADS * 4);
    unsigned* menc  = (unsigned*)alloc((size_t)NN * HEADS * 4);
    float*    denom = (float*)alloc((size_t)NN * HEADS * 4);
    float*    deg   = (float*)alloc((size_t)NN * 4);
    float*    normb = (float*)alloc((size_t)ETOT * 4);
    __bf16*   Wt    = (__bf16*)alloc((size_t)HDIM * HDIM * 2);
    float*    uemb  = (float*)alloc(64 * 4);
    float*    poolb = (float*)alloc(64 * 4);
    float*    vh    = (float*)alloc(256 * 4);
    float*    shv   = (float*)alloc(128 * 4);

    const int T = 256;
    auto blocks1d = [&](long n) { return (int)((n + T - 1) / T); };

    // GEMM helper: Y = Abf @ W (+bias), Abf bf16 [NN x 256], W fp32 [256 x Nout]
    auto gemm = [&](const __bf16* Abf, const float* W, const float* bias,
                    float* Y, int Nout) {
        k_wt<<<blocks1d((long)256 * Nout), T, 0, stream>>>(W, Wt, 256, Nout);
        if (Nout >= 64) {
            dim3 g((NN + 63) / 64, Nout / 64);
            k_gemm_bf16<4><<<g, 128, 0, stream>>>(Abf, Wt, bias, Y, NN, Nout);
        } else { // Nout == 32
            dim3 g((NN + 63) / 64, 1);
            k_gemm_bf16<2><<<g, 128, 0, stream>>>(Abf, Wt, bias, Y, NN, Nout);
        }
    };
    // BN helper (optional bf16 dual-store of the activation)
    auto bn = [&](const float* X, const float* g_, const float* b_, const float* res,
                  float* O, __bf16* Obf, int C, int mode) {
        k_colstats<<<C, 256, 0, stream>>>(X, NN, C, meanb, varb);
        long tot = (long)NN * C;
        k_bn_act<<<blocks1d(tot), T, 0, stream>>>(X, meanb, varb, g_, b_, res, O, Obf,
                                                  tot, C, mode);
    };
    // One GATv2 layer: bf16 input -> oOut (pre-BN; bias cb omitted: cancels in BN)
    auto gat = [&](const __bf16* hin, int li, float* oOut) {
        const float* Wl  = Wl_all  + (size_t)li * HDIM * HDIM;
        const float* bl  = bl_all  + (size_t)li * HDIM;
        const float* Wr  = Wr_all  + (size_t)li * HDIM * HDIM;
        const float* br  = br_all  + (size_t)li * HDIM;
        const float* att = att_all + (size_t)li * HEADS * CDIM;
        gemm(hin, Wl, bl, t0, HDIM);                                    // xl
        gemm(hin, Wr, br, t1, HDIM);                                    // xr
        k_fillu<<<blocks1d((long)NN * HEADS), T, 0, stream>>>(menc, ENC_NEG_INF, (long)NN * HEADS);
        k_fillf<<<blocks1d((long)NN * HEADS), T, 0, stream>>>(denom, 0.f, (long)NN * HEADS);
        k_edge_logit<<<blocks1d((long)ETOT * HEADS), T, 0, stream>>>(t0, t1, att, ei, logit, menc);
        k_edge_w<<<blocks1d((long)ETOT * HEADS), T, 0, stream>>>(logit, ei, menc, wbuf, denom);
        k_fillf<<<blocks1d((long)NN * HDIM), T, 0, stream>>>(oOut, 0.f, (long)NN * HDIM);
        k_gat_scatter<<<ETOT, HDIM, 0, stream>>>(t0, ei, wbuf, denom, oOut);
    };
    // GCN head: relu(BN(scatter(h @ W))) -> t2 [NN x F]  (bias cancels in BN)
    auto gcn_head = [&](const float* W, int F, const float* g_, const float* b_) {
        gemm(hbf, W, nullptr, t0, F);
        k_fillf<<<blocks1d((long)NN * F), T, 0, stream>>>(t1, 0.f, (long)NN * F);
        k_gcn_scatter<<<ETOT, F, 0, stream>>>(t0, ei, normb, t1, F);
        bn(t1, g_, b_, nullptr, t2, nullptr, F, /*relu*/1);
    };

    // ---- graph structure: degrees + symmetric norm (once)
    k_fillf<<<blocks1d(NN), T, 0, stream>>>(deg, 0.f, NN);
    k_deg <<<blocks1d(ETOT), T, 0, stream>>>(ei, deg);
    k_norm<<<blocks1d(ETOT), T, 0, stream>>>(ei, deg, normb);

    // ---- stem: h = mish(BN(x @ W_in)) -> hbuf (fp32) + hbf (bf16)
    k_in<<<blocks1d((long)NN * HDIM), T, 0, stream>>>(x, W_in, hbuf);
    bn(hbuf, g_in, be_in, nullptr, hbuf, hbf, HDIM, /*mish*/0);

    // ---- u embedding
    k_uemb<<<1, 64, 0, stream>>>(u, W_u, b_u, uemb);

    // ---- 8 residual GATv2 blocks
    for (int blk = 0; blk < NBLK; ++blk) {
        int l0 = blk * 2 + 0, l1 = blk * 2 + 1;
        // layer 0: o = mish(BN(gat(h)))  -> t2 (fp32) + tbf (bf16)
        gat(hbf, l0, t2);
        bn(t2, g_all + (size_t)l0 * HDIM, be_all + (size_t)l0 * HDIM, nullptr,
           t2, tbf, HDIM, 0);
        // layer 1: h = mish(BN(gat(o)) + h) -> hbuf + hbf
        gat(tbf, l1, t2);
        bn(t2, g_all + (size_t)l1 * HDIM, be_all + (size_t)l1 * HDIM, hbuf,
           hbuf, hbf, HDIM, 3);
    }

    // ---- policy head
    gcn_head(Wp, 128, gp, bep);
    k_node_dot<<<blocks1d(NN), T, 0, stream>>>(t2, 128, Wpf, bpf, out_policy, 0);

    // ---- value head
    gcn_head(Wv, 64, gv, bev);
    k_colsum<<<64, 256, 0, stream>>>(t2, NN, 64, poolb, 1.f / (float)NN);
    k_dense2_relu<<<1, 256, 0, stream>>>(poolb, 64, uemb, 64, Wv1, bv1, vh, 256);
    k_value_final<<<1, 1, 0, stream>>>(vh, Wv2, bv2, out_value);

    // ---- ownership head
    gcn_head(Wo, 64, go, beo);
    k_node_dot<<<blocks1d(NN), T, 0, stream>>>(t2, 64, Wof, bof, out_own, 1);

    // ---- score head
    gcn_head(Ws, 32, gs, bes);
    k_colsum<<<32, 256, 0, stream>>>(t2, NN, 32, poolb, 1.f / (float)NN);
    k_dense2_relu<<<1, 128, 0, stream>>>(poolb, 32, uemb, 64, Ws1, bs1, shv, 128);
    k_score_final<<<1, 1, 0, stream>>>(shv, Wsm, bsm, Wss, bss, out_smean, out_sstd);
}
